// IntNBitTableBatchedEmbeddingBagsCodegen_65369402245269
// MI455X (gfx1250) — matrix operand order repaired and checked by reading, not compiled, not run on hardware
//
#include <hip/hip_runtime.h>

// ---------------------------------------------------------------------------
// Table-batched INT8 embedding bag (FBGEMM IntNBit TBE) for MI455X (gfx1250).
//
// Roofline: ~250 MB moved -> ~11 us at 23.3 TB/s (memory floor).
//  * one wave32 per bag; 4 waves (128 thr) per block; rows land in LDS via
//    GLOBAL_LOAD_ASYNC_TO_LDS_B128 (ASYNCcnt), s_wait_asynccnt before use.
//  * pooling out[d] = sum_l (w*s)[l]*q[l][d] + sum_l w[l]*bias[l] done with
//    v_wmma_f32_16x16x32_f16; A = wEff broadcast along M, B = q tiles.
//  * B tiles are COLUMN-INTERLEAVED in groups of 4 (tile j covers columns
//    dg + 4n + j) so one ds_load_b32 per lane feeds element e of 4 B
//    matrices: 64 ds ops/bag instead of 256 -> LDS pipe (~3 us) drops below
//    the HBM floor; byte->f16 unpack via v_perm_b32 + dual-half v_cvt.
//  * per lane the 4 tile results are 4 consecutive output floats -> one
//    global_store_b128 per 64-column group.
// ---------------------------------------------------------------------------

#define T_TABLES 8
#define E_ROWS   250000
#define D_DIM    128
#define B_BAGS   4096
#define L_BAG    50

#define WAVES_PER_BLOCK 4
#define ROWS_PAD 64            // K padded to 64: rows 50..63 have wEff == 0

typedef __attribute__((ext_vector_type(16))) _Float16 v16h;
typedef __attribute__((ext_vector_type(8)))  float    v8f;

typedef __attribute__((address_space(3))) unsigned char lds_u8;

__device__ __forceinline__ void async_row16(const unsigned char* gp, unsigned char* lp) {
    // global_load_async_to_lds_b128: vdst = 32-bit LDS byte address,
    // vaddr = 64-bit global address, GV mode.
    unsigned long long ga = (unsigned long long)gp;
    unsigned la = (unsigned)(unsigned long long)(lds_u8*)lp;
    asm volatile("global_load_async_to_lds_b128 %0, %1, off"
                 :: "v"(la), "v"(ga) : "memory");
}

// unpack 4 packed u8 -> 4 f16 using v_perm_b32 ({0,b1,0,b0} / {0,b3,0,b2})
struct Half4 { _Float16 h0, h1, h2, h3; };

__device__ __forceinline__ Half4 unpack4(unsigned v) {
    unsigned p01 = __builtin_amdgcn_perm(v, v, 0x0c010c00u);
    unsigned p23 = __builtin_amdgcn_perm(v, v, 0x0c030c02u);
    Half4 r;
    r.h0 = (_Float16)(unsigned short)(p01);
    r.h1 = (_Float16)(unsigned short)(p01 >> 16);
    r.h2 = (_Float16)(unsigned short)(p23);
    r.h3 = (_Float16)(unsigned short)(p23 >> 16);
    return r;
}

__global__ __launch_bounds__(32 * WAVES_PER_BLOCK) void tbe_int8_fwd_kernel(
    const int*           __restrict__ indices,   // [T*B*L]
    const float*         __restrict__ psw,       // [T*B*L]
    const unsigned char* __restrict__ wq,        // [T,E,D] uint8
    const float*         __restrict__ scales,    // [T,E]
    const float*         __restrict__ biases,    // [T,E]
    float*               __restrict__ out)       // [B, T*D] (fp16-rounded fp32)
{
    __shared__ unsigned char shQ[WAVES_PER_BLOCK][ROWS_PAD * D_DIM]; // 32 KiB
    __shared__ _Float16      shW[WAVES_PER_BLOCK][ROWS_PAD];         // wEff

    const int lane   = threadIdx.x & 31;
    const int wv     = threadIdx.x >> 5;
    const int g      = blockIdx.x * WAVES_PER_BLOCK + wv;  // flat bag id (t*B+b)
    const int t      = g / B_BAGS;
    const int bag    = g - t * B_BAGS;
    const int laneHi = lane >> 4;       // 0: lanes 0-15, 1: lanes 16-31
    const int nm     = lane & 15;

    // ---- per-sample params: wEff = w*s (f16 into LDS), bagBias = sum w*bias
    const long ibase = (long)g * L_BAG;
    const long tbase = (long)t * E_ROWS;
    int rowIdx0, rowIdx1 = 0;
    float wb = 0.f;
    {
        int idx  = indices[ibase + lane];           // lane < 32 < L: always valid
        float w  = psw[ibase + lane];
        shW[wv][lane] = (_Float16)(w * scales[tbase + idx]);
        wb += w * biases[tbase + idx];
        rowIdx0 = idx;
    }
    {
        int l1 = lane + 32;
        if (l1 < L_BAG) {
            int idx  = indices[ibase + l1];
            float w  = psw[ibase + l1];
            shW[wv][l1] = (_Float16)(w * scales[tbase + idx]);
            wb += w * biases[tbase + idx];
            rowIdx1 = idx;
        } else {
            shW[wv][l1] = (_Float16)0.f;            // zero-pad K range [50,64)
        }
    }
    #pragma unroll
    for (int off = 16; off > 0; off >>= 1) wb += __shfl_down(wb, off, 32);
    const float bagBias = __shfl(wb, 0, 32);

    // ---- async gather: 13 x b128 -> 4 rows per issue (32 lanes x 16B each)
    const unsigned char* tablePtr = wq + (long)t * E_ROWS * D_DIM;
    const int sub = lane & 7;                        // 16B chunk within a row
    #pragma unroll
    for (int i = 0; i < 13; ++i) {                   // 52 row slots (>= L)
        int r   = i * 4 + (lane >> 3);               // row slot 0..51
        int idx = __shfl((i < 8) ? rowIdx0 : rowIdx1, r & 31, 32);
        async_row16(tablePtr + (long)idx * D_DIM + sub * 16,
                    &shQ[wv][r * D_DIM + sub * 16]);
    }

    // ---- build A (wEff broadcast along M) for both K-steps
    // A in contiguous K order (same as B): element e of lane holds K = 16*laneHi + e
    v16h a0, a1;
    #pragma unroll
    for (int e = 0; e < 16; ++e) {
        int K = e + (laneHi << 4);
        a0[e] = shW[wv][K];
        a1[e] = shW[wv][32 + K];
    }

    asm volatile("s_wait_asynccnt 0" ::: "memory");  // rows resident in LDS

    // ---- WMMA pooling: 2 column groups (64 dims) x 2 K-steps x 4 tiles
    // group tile j covers output columns dg + 4n + j  (n = 0..15)
    float* outBag = out + (long)bag * (T_TABLES * D_DIM) + t * D_DIM;
    const unsigned char* qBase = &shQ[wv][4 * nm];   // per-lane column base
    #pragma unroll
    for (int dg = 0; dg < D_DIM; dg += 64) {
        v8f c0 = {}, c1 = {}, c2 = {}, c3 = {};
        #pragma unroll
        for (int kstep = 0; kstep < 2; ++kstep) {
            const int kBase = kstep * 32;
            // B layout: element e of lane holds k = 16*laneHi + e, col = n
            v16h b0, b1, b2, b3;
            #pragma unroll
            for (int e = 0; e < 16; ++e) {
                int r = kBase + laneHi * 16 + e;     // row < 64, always in-bounds
                unsigned v = *(const unsigned*)(qBase + r * D_DIM + dg);
                Half4 u = unpack4(v);
                b0[e] = u.h0;
                b1[e] = u.h1;
                b2[e] = u.h2;
                b3[e] = u.h3;
            }
            const v16h a = kstep ? a1 : a0;
            c0 = __builtin_amdgcn_wmma_f32_16x16x32_f16(false, a, false, b0,
                                                        (short)0, c0, false, false);
            c1 = __builtin_amdgcn_wmma_f32_16x16x32_f16(false, a, false, b1,
                                                        (short)0, c1, false, false);
            c2 = __builtin_amdgcn_wmma_f32_16x16x32_f16(false, a, false, b2,
                                                        (short)0, c2, false, false);
            c3 = __builtin_amdgcn_wmma_f32_16x16x32_f16(false, a, false, b3,
                                                        (short)0, c3, false, false);
        }
        // D row M=0 (identical across M): VGPR0 of lanes 0-15 holds the result
        // for column n; lane n owns output dims dg+4n+{0,1,2,3} -> one b128.
        if (laneHi == 0) {
            float4 o;
            o.x = (float)(_Float16)(c0[0] + bagBias);  // fp16 rounding to match
            o.y = (float)(_Float16)(c1[0] + bagBias);  // reference fp16 output
            o.z = (float)(_Float16)(c2[0] + bagBias);
            o.w = (float)(_Float16)(c3[0] + bagBias);
            *(float4*)(outBag + dg + 4 * nm) = o;
        }
    }
}

extern "C" void kernel_launch(void* const* d_in, const int* in_sizes, int n_in,
                              void* d_out, int out_size, void* d_ws, size_t ws_size,
                              hipStream_t stream) {
    const int*           indices = (const int*)d_in[0];
    // d_in[1] = offsets: encodes fixed bag length L, not needed
    const float*         psw     = (const float*)d_in[2];
    const unsigned char* wq      = (const unsigned char*)d_in[3];
    const float*         scales  = (const float*)d_in[4];
    const float*         biases  = (const float*)d_in[5];
    float*               out     = (float*)d_out;

    dim3 grid((T_TABLES * B_BAGS) / WAVES_PER_BLOCK);  // 8192 blocks
    dim3 block(32 * WAVES_PER_BLOCK);                   // 4 waves of 32
    tbe_int8_fwd_kernel<<<grid, block, 0, stream>>>(indices, psw, wq,
                                                    scales, biases, out);
}
